// GAPTGN_56719338111556
// MI455X (gfx1250) — compile-verified, matrix operand order checked by reading, not run on hardware
//
#include <hip/hip_runtime.h>

// ---------------------------------------------------------------------------
// TGN-style fraud model for MI455X (gfx1250, wave32, WMMA).
// fp32 scatter-mean into L2-resident [N,416] buffer; f16-in/f32-acc WMMA for
// GRU, LSTM recurrence, and predictor GEMMs; static/price projections folded
// into pred_W1.
// ---------------------------------------------------------------------------

typedef __attribute__((ext_vector_type(16))) _Float16 v16h;
typedef __attribute__((ext_vector_type(8)))  float    v8f;
typedef __attribute__((ext_vector_type(4)))  float    v4f;

constexpr int E    = 200000;
constexpr int N    = 20000;
constexpr int H    = 128;
constexpr int EF   = 32;
constexpr int TT   = 14;
constexpr int LH   = 32;
constexpr int MSGW = 3 * H + EF;   // 416

static __device__ __forceinline__ v8f vzero() {
  v8f r;
#pragma unroll
  for (int i = 0; i < 8; ++i) r[i] = 0.f;
  return r;
}

static __device__ __forceinline__ v16h hzero() {
  v16h r;
#pragma unroll
  for (int i = 0; i < 16; ++i) r[i] = (_Float16)0.f;
  return r;
}

// Build a v16h A/B fragment for wmma_f32_16x16x32_f16.
// p0 -> 8 consecutive f32 covering K = k0 + hi*8 .. +7
// p1 -> 8 consecutive f32 covering K = k0 + 16 + hi*8 .. +7
// (16-B aligned; works for global and LDS pointers)
static __device__ __forceinline__ v16h pack16(const float* p0, const float* p1) {
  v4f a0 = *(const v4f*)(p0);
  v4f a1 = *(const v4f*)(p0 + 4);
  v4f b0 = *(const v4f*)(p1);
  v4f b1 = *(const v4f*)(p1 + 4);
  v16h r;
#pragma unroll
  for (int i = 0; i < 4; ++i) {
    r[i]      = (_Float16)a0[i];
    r[4 + i]  = (_Float16)a1[i];
    r[8 + i]  = (_Float16)b0[i];
    r[12 + i] = (_Float16)b1[i];
  }
  return r;
}

static __device__ __forceinline__ v8f wmma16(v16h a, v16h b, v8f c) {
  return __builtin_amdgcn_wmma_f32_16x16x32_f16(false, a, false, b, (short)0, c,
                                                false, false);
}

static __device__ __forceinline__ float sigf(float x) {
  return 1.f / (1.f + __expf(-x));
}

// ---------------------------------------------------------------------------
// 1) zero scratch (agg + cnt are accumulated via atomics -> must re-zero)
// ---------------------------------------------------------------------------
__global__ void k_zero(float* __restrict__ p, int n) {
  for (int i = blockIdx.x * blockDim.x + threadIdx.x; i < n;
       i += gridDim.x * blockDim.x)
    p[i] = 0.f;
}

// ---------------------------------------------------------------------------
// 2) per-edge message build + scatter-mean numerator.  One wave per message
//    (2E messages).  Lane j covers 13 strided positions of the 416-dim msg.
// ---------------------------------------------------------------------------
__global__ void k_scatter(const int* __restrict__ src, const int* __restrict__ dst,
                          const float* __restrict__ tt, const float* __restrict__ msg,
                          const float* __restrict__ memory,
                          const float* __restrict__ last_update,
                          const float* __restrict__ time_w,
                          const float* __restrict__ time_b,
                          float* __restrict__ agg, float* __restrict__ cnt) {
  int gid  = blockIdx.x * blockDim.x + threadIdx.x;
  int m    = gid >> 5;
  int lane = gid & 31;
  if (m >= 2 * E) return;
  int e = (m < E) ? m : (m - E);
  int s = src[e], d = dst[e];
  int target = (m < E) ? d : s;   // msg_to_dst first, then msg_to_src
  int first  = (m < E) ? s : d;   // slot0 = memory[first]
  int second = (m < E) ? d : s;   // slot1 = memory[second]
  float dt   = tt[e] - last_update[target];
  float* row = agg + (size_t)target * MSGW;
#pragma unroll
  for (int i = 0; i < 13; ++i) {
    int p = lane + 32 * i;
    float v;
    if (p < H)               v = memory[(size_t)first * H + p];
    else if (p < 2 * H)      v = memory[(size_t)second * H + (p - H)];
    else if (p < 2 * H + EF) v = msg[(size_t)e * EF + (p - 2 * H)];
    else {
      int q = p - (2 * H + EF);
      v = __cosf(dt * time_w[q] + time_b[q]);
    }
    atomicAdd(row + p, v);
  }
  if (lane == 0) atomicAdd(cnt + target, 1.f);
}

// ---------------------------------------------------------------------------
// 3) agg /= max(cnt,1)
// ---------------------------------------------------------------------------
__global__ void k_divmean(float* __restrict__ agg, const float* __restrict__ cnt) {
  int total = N * MSGW;
  for (int i = blockIdx.x * blockDim.x + threadIdx.x; i < total;
       i += gridDim.x * blockDim.x) {
    int node = i / MSGW;
    agg[i] = agg[i] / fmaxf(cnt[node], 1.f);
  }
}

// ---------------------------------------------------------------------------
// 4) GRU memory update, WMMA.  One wave = 16 nodes x 32 units.
//    gi = agg@Wih.T (K=416), gh = memory@Whh.T (K=128).
//    r/z accumulators take both GEMMs; n-gate kept split (needs r*h_n).
// ---------------------------------------------------------------------------
__global__ void __launch_bounds__(128) k_gru(
    const float* __restrict__ aggm, const float* __restrict__ cnt,
    const float* __restrict__ memory, const float* __restrict__ Wih,
    const float* __restrict__ Whh, const float* __restrict__ bih,
    const float* __restrict__ bhh, float* __restrict__ mem_new) {
  int wg   = blockIdx.x * (blockDim.x >> 5) + (threadIdx.x >> 5);
  int lane = threadIdx.x & 31;
  int nt = wg >> 2, ut = wg & 3;
  if (nt >= N / 16) return;
  int col = lane & 15, hi = lane >> 4;
  int arow = nt * 16 + col;       // A-fragment row owned by this lane
  int u0 = ut * 32;

  v8f ar0 = vzero(), ar1 = vzero(), az0 = vzero(), az1 = vzero();
  v8f ain0 = vzero(), ain1 = vzero(), ahn0 = vzero(), ahn1 = vzero();

  // phase 1: agg @ Wih.T   (13 K-steps of 32)
  for (int kb = 0; kb < 13; ++kb) {
    int k0 = kb * 32;
    const float* ap = aggm + (size_t)arow * MSGW + k0;
    v16h a = pack16(ap + hi * 8, ap + 16 + hi * 8);
    const float* wr0 = Wih + (size_t)(u0 + col) * MSGW + k0;
    const float* wr1 = Wih + (size_t)(u0 + 16 + col) * MSGW + k0;
    const float* wz0 = Wih + (size_t)(H + u0 + col) * MSGW + k0;
    const float* wz1 = Wih + (size_t)(H + u0 + 16 + col) * MSGW + k0;
    const float* wn0 = Wih + (size_t)(2 * H + u0 + col) * MSGW + k0;
    const float* wn1 = Wih + (size_t)(2 * H + u0 + 16 + col) * MSGW + k0;
    ar0  = wmma16(a, pack16(wr0 + hi * 8, wr0 + 16 + hi * 8), ar0);
    ar1  = wmma16(a, pack16(wr1 + hi * 8, wr1 + 16 + hi * 8), ar1);
    az0  = wmma16(a, pack16(wz0 + hi * 8, wz0 + 16 + hi * 8), az0);
    az1  = wmma16(a, pack16(wz1 + hi * 8, wz1 + 16 + hi * 8), az1);
    ain0 = wmma16(a, pack16(wn0 + hi * 8, wn0 + 16 + hi * 8), ain0);
    ain1 = wmma16(a, pack16(wn1 + hi * 8, wn1 + 16 + hi * 8), ain1);
  }
  // phase 2: memory @ Whh.T  (4 K-steps of 32)
  for (int kb = 0; kb < 4; ++kb) {
    int k0 = kb * 32;
    const float* ap = memory + (size_t)arow * H + k0;
    v16h a = pack16(ap + hi * 8, ap + 16 + hi * 8);
    const float* wr0 = Whh + (size_t)(u0 + col) * H + k0;
    const float* wr1 = Whh + (size_t)(u0 + 16 + col) * H + k0;
    const float* wz0 = Whh + (size_t)(H + u0 + col) * H + k0;
    const float* wz1 = Whh + (size_t)(H + u0 + 16 + col) * H + k0;
    const float* wn0 = Whh + (size_t)(2 * H + u0 + col) * H + k0;
    const float* wn1 = Whh + (size_t)(2 * H + u0 + 16 + col) * H + k0;
    ar0  = wmma16(a, pack16(wr0 + hi * 8, wr0 + 16 + hi * 8), ar0);
    ar1  = wmma16(a, pack16(wr1 + hi * 8, wr1 + 16 + hi * 8), ar1);
    az0  = wmma16(a, pack16(wz0 + hi * 8, wz0 + 16 + hi * 8), az0);
    az1  = wmma16(a, pack16(wz1 + hi * 8, wz1 + 16 + hi * 8), az1);
    ahn0 = wmma16(a, pack16(wn0 + hi * 8, wn0 + 16 + hi * 8), ahn0);
    ahn1 = wmma16(a, pack16(wn1 + hi * 8, wn1 + 16 + hi * 8), ahn1);
  }
  // epilogue in C layout: lane -> column, vgpr -> row (m = vg + hi*8)
#pragma unroll
  for (int vg = 0; vg < 8; ++vg) {
    int m = vg + hi * 8;
    int node = nt * 16 + m;
    float cv = cnt[node];
#pragma unroll
    for (int s = 0; s < 2; ++s) {
      int unit = u0 + s * 16 + col;
      float gr  = (s ? ar1[vg] : ar0[vg]) + bih[unit] + bhh[unit];
      float gz  = (s ? az1[vg] : az0[vg]) + bih[H + unit] + bhh[H + unit];
      float gin = (s ? ain1[vg] : ain0[vg]) + bih[2 * H + unit];
      float ghn = (s ? ahn1[vg] : ahn0[vg]) + bhh[2 * H + unit];
      float r  = sigf(gr);
      float z  = sigf(gz);
      float nn = tanhf(gin + r * ghn);
      float mv = memory[(size_t)node * H + unit];
      float hn = (1.f - z) * nn + z * mv;
      mem_new[(size_t)node * H + unit] = (cv > 0.f) ? hn : mv;
    }
  }
}

// ---------------------------------------------------------------------------
// 5) fold static_W/price_W/b into pred_W1 space:
//    WsT[n][k] = sum_j static_W[j][k] * W1[n][j]
//    WpT[n][k] = sum_j price_W[j][k] * (W1[n][128+j] + W1[n][256+j])
//    biasf[n]  = b1[n] + static_b@W1a[n] + price_b@(W1b[n]+W1c[n])
// ---------------------------------------------------------------------------
__global__ void k_fuse(const float* __restrict__ static_W,
                       const float* __restrict__ static_b,
                       const float* __restrict__ price_W,
                       const float* __restrict__ price_b,
                       const float* __restrict__ W1, const float* __restrict__ b1,
                       float* __restrict__ WsT, float* __restrict__ WpT,
                       float* __restrict__ biasf) {
  int i = blockIdx.x * blockDim.x + threadIdx.x;
  if (i < 64 * LH) {
    int n = i >> 5, k = i & 31;
    float s = 0.f, p = 0.f;
    for (int j = 0; j < H; ++j) {
      s += static_W[j * LH + k] * W1[(size_t)n * 384 + j];
      p += price_W[j * LH + k] *
           (W1[(size_t)n * 384 + H + j] + W1[(size_t)n * 384 + 2 * H + j]);
    }
    WsT[i] = s;
    WpT[i] = p;
  } else if (i < 64 * LH + 64) {
    int n = i - 64 * LH;
    float b = b1[n];
    for (int j = 0; j < H; ++j) {
      b += static_b[j] * W1[(size_t)n * 384 + j] +
           price_b[j] *
               (W1[(size_t)n * 384 + H + j] + W1[(size_t)n * 384 + 2 * H + j]);
    }
    biasf[n] = b;
  }
}

// ---------------------------------------------------------------------------
// 6) price LSTM via WMMA.  One wave = 16 edges.
//    h state kept as a 16x32 f16 A-fragment; Whh preloaded once as 8 B
//    fragments (gate tiles i0,i1,f0,f1,g0,g1,o0,o1).  Per step:
//      acc[jt] = bias + x*Wih (seeded in C layout), then 8 WMMAs for h@Whh.T,
//      gate math elementwise in C layout (c state = 2 v8f C tiles),
//      h rebuilt into A layout via a conflict-free stride-36 LDS transpose.
// ---------------------------------------------------------------------------
__global__ void __launch_bounds__(128) k_lstm(
    const float* __restrict__ price_seq, const float* __restrict__ Wih,
    const float* __restrict__ Whh, const float* __restrict__ bih,
    const float* __restrict__ bhh, float* __restrict__ h_last) {
  __shared__ __align__(16) float sHT[4 * 16 * 36];
  int wslot = threadIdx.x >> 5;
  int lane  = threadIdx.x & 31;
  int wg    = blockIdx.x * (blockDim.x >> 5) + wslot;
  if (wg >= E / 16) return;
  int col = lane & 15, hi = lane >> 4;
  float* tile = sHT + wslot * (16 * 36);

  // preload recurrence weights: B fragment jt covers gate rows jt*16 + col
  v16h bf[8];
  float wx[8], bs[8];
#pragma unroll
  for (int jt = 0; jt < 8; ++jt) {
    int g = jt * 16 + col;
    const float* wp = Whh + (size_t)g * LH;
    bf[jt] = pack16(wp + hi * 8, wp + 16 + hi * 8);
    wx[jt] = Wih[g];
    bs[jt] = bih[g] + bhh[g];
  }

  v8f c0 = vzero(), c1 = vzero();
  v16h hA = hzero();
  int e_lane = wg * 16 + col;   // edge whose x this lane loads

  for (int t = 0; t < TT; ++t) {
    float xl = price_seq[(size_t)e_lane * TT + t];
    float xv[8];
#pragma unroll
    for (int vg = 0; vg < 8; ++vg) xv[vg] = __shfl(xl, vg + hi * 8, 32);

    v8f acc[8];
#pragma unroll
    for (int jt = 0; jt < 8; ++jt) {
#pragma unroll
      for (int vg = 0; vg < 8; ++vg) acc[jt][vg] = bs[jt] + xv[vg] * wx[jt];
      acc[jt] = wmma16(hA, bf[jt], acc[jt]);
    }

    // gate math in C layout; subtile s covers units s*16 + col
#pragma unroll
    for (int s = 0; s < 2; ++s) {
      v8f cc = s ? c1 : c0;
#pragma unroll
      for (int vg = 0; vg < 8; ++vg) {
        float iv = sigf(acc[0 + s][vg]);
        float fv = sigf(acc[2 + s][vg]);
        float gv = tanhf(acc[4 + s][vg]);
        float ov = sigf(acc[6 + s][vg]);
        float cv = fv * cc[vg] + iv * gv;
        cc[vg] = cv;
        float hv = ov * tanhf(cv);
        tile[(vg + hi * 8) * 36 + s * 16 + col] = hv;
      }
      if (s) c1 = cc; else c0 = cc;
    }
    // transpose h back to A layout (same-wave LDS ops are in-order)
    __builtin_amdgcn_wave_barrier();
    const float* rp = tile + (size_t)col * 36;
    hA = pack16(rp + hi * 8, rp + 16 + hi * 8);
    __builtin_amdgcn_wave_barrier();
  }
  // final hidden state (f32) for the predictor, contiguous 16-float runs
  const float* rp = tile + (size_t)col * 36 + hi * 16;
  float* op = h_last + (size_t)(wg * 16 + col) * LH + hi * 16;
#pragma unroll
  for (int j = 0; j < 16; ++j) op[j] = rp[j];
}

// ---------------------------------------------------------------------------
// 7) fused predictor, WMMA.  One wave = 16 edges x 64 hidden units.
//    hid = relu(mem_src@W1a.T + mem_dst@W1b.T + feat32@Ws' + h32@Wp' + biasf)
//    out = hid @ W2.T + b2   (folded in via 16-lane shfl_xor reduction)
// ---------------------------------------------------------------------------
__global__ void __launch_bounds__(128) k_pred(
    const int* __restrict__ src, const int* __restrict__ dst,
    const int* __restrict__ xs, const float* __restrict__ mem_new,
    const float* __restrict__ h_last, const float* __restrict__ party_emb,
    const float* __restrict__ state_emb, const float* __restrict__ W1,
    const float* __restrict__ WsT, const float* __restrict__ WpT,
    const float* __restrict__ biasf, const float* __restrict__ W2,
    const float* __restrict__ b2, float* __restrict__ out) {
  int wg   = blockIdx.x * (blockDim.x >> 5) + (threadIdx.x >> 5);
  int lane = threadIdx.x & 31;
  if (wg >= E / 16) return;
  int col = lane & 15, hi = lane >> 4;
  int e  = wg * 16 + col;
  int se = src[e], de = dst[e];

  v8f a0 = vzero(), a1 = vzero(), a2 = vzero(), a3 = vzero();
  const float* w1r0 = W1 + (size_t)(col) * 384;
  const float* w1r1 = W1 + (size_t)(16 + col) * 384;
  const float* w1r2 = W1 + (size_t)(32 + col) * 384;
  const float* w1r3 = W1 + (size_t)(48 + col) * 384;

  // mem_src against W1 columns [0,128)
  for (int kb = 0; kb < 4; ++kb) {
    int k0 = kb * 32;
    const float* ap = mem_new + (size_t)se * H + k0;
    v16h a = pack16(ap + hi * 8, ap + 16 + hi * 8);
    a0 = wmma16(a, pack16(w1r0 + k0 + hi * 8, w1r0 + k0 + 16 + hi * 8), a0);
    a1 = wmma16(a, pack16(w1r1 + k0 + hi * 8, w1r1 + k0 + 16 + hi * 8), a1);
    a2 = wmma16(a, pack16(w1r2 + k0 + hi * 8, w1r2 + k0 + 16 + hi * 8), a2);
    a3 = wmma16(a, pack16(w1r3 + k0 + hi * 8, w1r3 + k0 + 16 + hi * 8), a3);
  }
  // mem_dst against W1 columns [128,256)
  for (int kb = 0; kb < 4; ++kb) {
    int k0 = kb * 32;
    const float* ap = mem_new + (size_t)de * H + k0;
    v16h a = pack16(ap + hi * 8, ap + 16 + hi * 8);
    a0 = wmma16(a, pack16(w1r0 + H + k0 + hi * 8, w1r0 + H + k0 + 16 + hi * 8), a0);
    a1 = wmma16(a, pack16(w1r1 + H + k0 + hi * 8, w1r1 + H + k0 + 16 + hi * 8), a1);
    a2 = wmma16(a, pack16(w1r2 + H + k0 + hi * 8, w1r2 + H + k0 + 16 + hi * 8), a2);
    a3 = wmma16(a, pack16(w1r3 + H + k0 + hi * 8, w1r3 + H + k0 + 16 + hi * 8), a3);
  }
  // static 32-d feature (party||state) against fused Ws'
  {
    int pp = xs[(size_t)se * 2 + 0];
    int st = xs[(size_t)se * 2 + 1];
    v16h a = pack16(party_emb + (size_t)pp * 16 + hi * 8,
                    state_emb + (size_t)st * 16 + hi * 8);
    a0 = wmma16(a, pack16(WsT + (size_t)(col) * 32 + hi * 8,
                          WsT + (size_t)(col) * 32 + 16 + hi * 8), a0);
    a1 = wmma16(a, pack16(WsT + (size_t)(16 + col) * 32 + hi * 8,
                          WsT + (size_t)(16 + col) * 32 + 16 + hi * 8), a1);
    a2 = wmma16(a, pack16(WsT + (size_t)(32 + col) * 32 + hi * 8,
                          WsT + (size_t)(32 + col) * 32 + 16 + hi * 8), a2);
    a3 = wmma16(a, pack16(WsT + (size_t)(48 + col) * 32 + hi * 8,
                          WsT + (size_t)(48 + col) * 32 + 16 + hi * 8), a3);
  }
  // LSTM hidden (32) against fused Wp'
  {
    const float* hp = h_last + (size_t)e * LH;
    v16h a = pack16(hp + hi * 8, hp + 16 + hi * 8);
    a0 = wmma16(a, pack16(WpT + (size_t)(col) * 32 + hi * 8,
                          WpT + (size_t)(col) * 32 + 16 + hi * 8), a0);
    a1 = wmma16(a, pack16(WpT + (size_t)(16 + col) * 32 + hi * 8,
                          WpT + (size_t)(16 + col) * 32 + 16 + hi * 8), a1);
    a2 = wmma16(a, pack16(WpT + (size_t)(32 + col) * 32 + hi * 8,
                          WpT + (size_t)(32 + col) * 32 + 16 + hi * 8), a2);
    a3 = wmma16(a, pack16(WpT + (size_t)(48 + col) * 32 + hi * 8,
                          WpT + (size_t)(48 + col) * 32 + 16 + hi * 8), a3);
  }

  float b2v = b2[0];
  float bf0 = biasf[col],      w20 = W2[col];
  float bf1 = biasf[16 + col], w21 = W2[16 + col];
  float bf2 = biasf[32 + col], w22 = W2[32 + col];
  float bf3 = biasf[48 + col], w23 = W2[48 + col];
  int ebase = wg * 16 + hi * 8;
#pragma unroll
  for (int vg = 0; vg < 8; ++vg) {
    float s = fmaxf(a0[vg] + bf0, 0.f) * w20 + fmaxf(a1[vg] + bf1, 0.f) * w21 +
              fmaxf(a2[vg] + bf2, 0.f) * w22 + fmaxf(a3[vg] + bf3, 0.f) * w23;
#pragma unroll
    for (int off = 1; off < 16; off <<= 1) s += __shfl_xor(s, off, 32);
    if (col == 0) out[ebase + vg] = s + b2v;
  }
}

// ---------------------------------------------------------------------------
extern "C" void kernel_launch(void* const* d_in, const int* in_sizes, int n_in,
                              void* d_out, int out_size, void* d_ws,
                              size_t ws_size, hipStream_t stream) {
  (void)in_sizes; (void)n_in; (void)out_size; (void)ws_size;
  const int*   src         = (const int*)d_in[0];
  const int*   dst         = (const int*)d_in[1];
  const float* tt          = (const float*)d_in[2];
  const float* msg         = (const float*)d_in[3];
  const float* price_seq   = (const float*)d_in[4];
  /* d_in[5] trade_t unused by reference */
  const int*   x_static    = (const int*)d_in[6];
  const float* memory      = (const float*)d_in[7];
  const float* last_update = (const float*)d_in[8];
  const float* time_w      = (const float*)d_in[9];
  const float* time_b      = (const float*)d_in[10];
  const float* gru_Wih     = (const float*)d_in[11];
  const float* gru_Whh     = (const float*)d_in[12];
  const float* gru_bih     = (const float*)d_in[13];
  const float* gru_bhh     = (const float*)d_in[14];
  const float* party_emb   = (const float*)d_in[15];
  const float* state_emb   = (const float*)d_in[16];
  const float* static_W    = (const float*)d_in[17];
  const float* static_b    = (const float*)d_in[18];
  const float* lstm_Wih    = (const float*)d_in[19];
  const float* lstm_Whh    = (const float*)d_in[20];
  const float* lstm_bih    = (const float*)d_in[21];
  const float* lstm_bhh    = (const float*)d_in[22];
  const float* price_W     = (const float*)d_in[23];
  const float* price_b     = (const float*)d_in[24];
  const float* pred_W1     = (const float*)d_in[25];
  const float* pred_b1     = (const float*)d_in[26];
  const float* pred_W2     = (const float*)d_in[27];
  const float* pred_b2     = (const float*)d_in[28];
  float* out = (float*)d_out;

  // workspace layout (floats): ~69.3 MB total
  float* ws      = (float*)d_ws;
  float* agg     = ws;                           // N*416
  float* cnt     = agg + (size_t)N * MSGW;       // N
  float* mem_new = cnt + N;                      // N*128
  float* h_last  = mem_new + (size_t)N * H;      // E*32
  float* WsT     = h_last + (size_t)E * LH;      // 64*32
  float* WpT     = WsT + 64 * LH;                // 64*32
  float* biasf   = WpT + 64 * LH;                // 64

  k_zero<<<2048, 256, 0, stream>>>(agg, N * MSGW + N);
  k_scatter<<<(2 * E) / 8, 256, 0, stream>>>(src, dst, tt, msg, memory,
                                             last_update, time_w, time_b, agg,
                                             cnt);
  k_divmean<<<2048, 256, 0, stream>>>(agg, cnt);
  k_gru<<<(N / 16) * 4 / 4, 128, 0, stream>>>(agg, cnt, memory, gru_Wih,
                                              gru_Whh, gru_bih, gru_bhh,
                                              mem_new);
  k_fuse<<<(64 * LH + 64 + 255) / 256, 256, 0, stream>>>(
      static_W, static_b, price_W, price_b, pred_W1, pred_b1, WsT, WpT, biasf);
  k_lstm<<<(E / 16) / 4, 128, 0, stream>>>(price_seq, lstm_Wih, lstm_Whh,
                                           lstm_bih, lstm_bhh, h_last);
  k_pred<<<(E / 16) / 4, 128, 0, stream>>>(src, dst, x_static, mem_new, h_last,
                                           party_emb, state_emb, pred_W1, WsT,
                                           WpT, biasf, pred_W2, pred_b2, out);
}
